// NexusGraph_17291538333804
// MI455X (gfx1250) — matrix operand order repaired and checked by reading, not compiled, not run on hardware
//
#include <hip/hip_runtime.h>
#include <cstdint>
#include <cstddef>

// ---------------------------------------------------------------------------
// 2-layer TransformerConv GNN for MI455X (gfx1250, wave32, WMMA).
// Dense projections: v_wmma_f32_16x16x32_f16 with double-buffered LDS tiles
// staged via global_load_async_to_lds_b128 (ASYNCcnt partial waits overlap
// copy-in of tile t+1 with WMMA on tile t). Weights are pre-transposed so
// both A and B tiles are contiguous copies. Edge softmax/scatter phases are
// wave32 shuffles + global float atomics (HBM gather/scatter bound).
// ---------------------------------------------------------------------------

typedef _Float16 half16 __attribute__((ext_vector_type(16)));
typedef float    float8 __attribute__((ext_vector_type(8)));

static constexpr int N_NODES = 50000;
static constexpr int N_EDGES = 640000;
static constexpr int DIN = 256;
static constexpr int DE  = 64;
static constexpr int H1  = 128;
static constexpr int H2  = 64;

// ---------------- f32 -> f16 convert ----------------
__global__ void cvt_f32_f16(const float* __restrict__ s, _Float16* __restrict__ d, int n) {
    int i = blockIdx.x * blockDim.x + threadIdx.x;
    if (i < n) d[i] = (_Float16)s[i];
}

// ---------------- f32 -> f16 convert with transpose: s[K][N] -> d[N][K] -----
__global__ void cvt_f32_f16_T(const float* __restrict__ s, _Float16* __restrict__ d,
                              int K, int N) {
    int i = blockIdx.x * blockDim.x + threadIdx.x;
    if (i < K * N) {
        int k = i / N, n = i % N;
        d[(size_t)n * K + k] = (_Float16)s[i];
    }
}

// ---------------- ReLU + f16 convert ----------------
__global__ void relu_cvt_f16(const float* __restrict__ s, _Float16* __restrict__ d, int n) {
    int i = blockIdx.x * blockDim.x + threadIdx.x;
    if (i < n) { float v = s[i]; d[i] = (_Float16)(v > 0.0f ? v : 0.0f); }
}

// ---------------- segment-softmax init ----------------
__global__ void attn_init(unsigned* __restrict__ mEnc, float* __restrict__ den, int n) {
    int i = blockIdx.x * blockDim.x + threadIdx.x;
    if (i < n) { mEnc[i] = 0u; den[i] = 0.0f; }
}

// order-preserving float<->uint encoding (for atomicMax-based segment max)
__device__ __forceinline__ unsigned ford(float f) {
    unsigned u = __float_as_uint(f);
    return (u & 0x80000000u) ? ~u : (u | 0x80000000u);
}
__device__ __forceinline__ float funord(unsigned u) {
    u = (u & 0x80000000u) ? (u & 0x7FFFFFFFu) : ~u;
    return __uint_as_float(u);
}

// gfx1250 async global->LDS copy of 16 bytes (per active lane), ASYNCcnt-tracked
__device__ __forceinline__ void async_copy_b128(const _Float16* gptr, uint32_t lds_off) {
    asm volatile("global_load_async_to_lds_b128 %0, %1, off"
                 :: "v"(lds_off), "v"((uint64_t)(uintptr_t)gptr)
                 : "memory");
}
__device__ __forceinline__ void wait_asynccnt0() {
    asm volatile("s_wait_asynccnt 0x0" ::: "memory");
}
// allow 4 in-flight async ops (the next tile's copies) while draining older ones
__device__ __forceinline__ void wait_asynccnt4() {
    asm volatile("s_wait_asynccnt 0x4" ::: "memory");
}

// ---------------- WMMA GEMM: C[M,N] = A[M,K](f16) * BT[N,K](f16) + bias -----
// B is supplied TRANSPOSED ([N][K] row-major) so both LDS tiles are plain
// contiguous copies -> async-to-LDS staging, double buffered.
// Block: 128 threads = 4 waves. Block tile 64x64, K-step 32.
// Wave w computes rows [16w,16w+16) x all 64 cols (4 wmma 16x16 tiles).
__global__ __launch_bounds__(128)
void gemm_f16_wmma(const _Float16* __restrict__ A, const _Float16* __restrict__ BT,
                   const float* __restrict__ bias,
                   float* __restrict__ C, _Float16* __restrict__ Ch,
                   int M, int N, int K)
{
    __shared__ _Float16 Alds[2][64][40] __attribute__((aligned(16)));  // [m][k], 80B rows
    __shared__ _Float16 Blds[2][64][40] __attribute__((aligned(16)));  // [n][k]

    const int tid   = threadIdx.x;
    const int lane  = tid & 31;
    const int wave  = tid >> 5;
    const int mBase = blockIdx.x * 64;
    const int nBase = blockIdx.y * 64;
    const int l = lane & 15;   // N (or M) index within 16
    const int g = lane >> 4;   // lane half

    const bool fullA = (mBase + 64 <= M);   // uniform over the block

    float8 zero = {};
    float8 acc[4] = { zero, zero, zero, zero };

    // stage one 64x32 A tile + 64x32 B tile into LDS buffer `buf`
    auto stage = [&](int kb, int buf) {
        if (fullA) {
            #pragma unroll
            for (int it = 0; it < 2; ++it) {
                int c   = tid + it * 128;           // 256 x 16B chunks
                int row = c >> 2;
                int col = (c & 3) * 8;
                const _Float16* gp = &A[(size_t)(mBase + row) * K + kb + col];
                async_copy_b128(gp, (uint32_t)(uintptr_t)&Alds[buf][row][col]);
            }
        } else {
            #pragma unroll
            for (int it = 0; it < 2; ++it) {
                int c   = tid + it * 128;
                int row = c >> 2;
                int col = (c & 3) * 8;
                float4 v;
                int grow = mBase + row;
                if (grow < M) v = *reinterpret_cast<const float4*>(&A[(size_t)grow * K + kb + col]);
                else          v = make_float4(0.f, 0.f, 0.f, 0.f);
                *reinterpret_cast<float4*>(&Alds[buf][row][col]) = v;
            }
        }
        #pragma unroll
        for (int it = 0; it < 2; ++it) {
            int c   = tid + it * 128;
            int row = c >> 2;
            int col = (c & 3) * 8;
            const _Float16* gp = &BT[(size_t)(nBase + row) * K + kb + col];
            async_copy_b128(gp, (uint32_t)(uintptr_t)&Blds[buf][row][col]);
        }
    };

    // 4 WMMAs on buffer `buf`
    auto compute = [&](int buf) {
        // A fragment: dword d -> K = 2d + 16g (same k order as B)
        union { uint32_t u[8]; half16 v; } af;
        const int arow = 16 * wave + l;
        #pragma unroll
        for (int d = 0; d < 8; ++d) {
            int kb = 2 * d + 16 * g;
            af.u[d] = *reinterpret_cast<const uint32_t*>(&Alds[buf][arow][kb]);
        }
        #pragma unroll
        for (int nt = 0; nt < 4; ++nt) {
            // B fragment: contiguous k order (same as A): lanes0-15 K=0..15, lanes16-31 K=16..31 -> K = 2d + 16g
            union { uint32_t u[8]; half16 v; } bf;
            const int bcol = 16 * nt + l;
            #pragma unroll
            for (int d = 0; d < 8; ++d) {
                int kb = 2 * d + 16 * g;
                bf.u[d] = *reinterpret_cast<const uint32_t*>(&Blds[buf][bcol][kb]);
            }
            acc[nt] = __builtin_amdgcn_wmma_f32_16x16x32_f16(
                false, af.v, false, bf.v, (short)0, acc[nt], false, false);
        }
    };

    const int nK = K >> 5;                 // K is a multiple of 32
    stage(0, 0);                           // prologue
    for (int kt = 0; kt < nK; ++kt) {
        const int  cur  = kt & 1;
        const bool more = (kt + 1 < nK);
        if (more) {
            if (fullA && (kt + 2 < nK)) {
                // warm L2 two stages ahead of the async copy engine
                int c   = tid;
                int row = c >> 2;
                int col = (c & 3) * 8;
                __builtin_prefetch(&A[(size_t)(mBase + row) * K + (kt + 2) * 32 + col], 0, 3);
            }
            stage((kt + 1) * 32, cur ^ 1); // issue next tile's async copies
        }
        // async loads complete in order: <=4 outstanding means tile `cur` landed
        if (fullA && more) wait_asynccnt4();
        else               wait_asynccnt0();
        __syncthreads();
        compute(cur);
        __syncthreads();                   // readers done before buffer reuse
    }

    // C layout: element r of v8f -> M = r + 8g ; N = l
    #pragma unroll
    for (int nt = 0; nt < 4; ++nt) {
        int gcol = nBase + 16 * nt + l;
        float bv = bias ? bias[gcol] : 0.0f;
        #pragma unroll
        for (int r = 0; r < 8; ++r) {
            int grow = mBase + 16 * wave + r + 8 * g;
            if (grow < M) {
                float v = acc[nt][r] + bv;
                if (Ch) Ch[(size_t)grow * N + gcol] = (_Float16)v;
                else    C [(size_t)grow * N + gcol] = v;
            }
        }
    }
}

// ---------------- attention phase 1: alpha + segment max --------------------
// one wave32 per edge: alpha = dot(q[dst], k[src] + e) / sqrt(C)
__global__ __launch_bounds__(256)
void attn_alpha(const float* __restrict__ q, const float* __restrict__ k,
                const _Float16* __restrict__ e,
                const int* __restrict__ src, const int* __restrict__ dst,
                float* __restrict__ alpha, unsigned* __restrict__ mEnc,
                int E, int C, float rsqC)
{
    int edge = blockIdx.x * 8 + (threadIdx.x >> 5);
    int lane = threadIdx.x & 31;
    if (edge >= E) return;
    int s = src[edge], d = dst[edge];
    float acc = 0.0f;
    for (int c = lane; c < C; c += 32)
        acc += q[(size_t)d * C + c] * (k[(size_t)s * C + c] + (float)e[(size_t)edge * C + c]);
    for (int o = 16; o > 0; o >>= 1) acc += __shfl_down(acc, o, 32);
    if (lane == 0) {
        float a = acc * rsqC;
        alpha[edge] = a;
        atomicMax(&mEnc[d], ford(a));
    }
}

// ---------------- attention phase 2: exp + segment sum ----------------------
__global__ void attn_exp(float* __restrict__ alpha, const unsigned* __restrict__ mEnc,
                         float* __restrict__ den, const int* __restrict__ dst, int E)
{
    int edge = blockIdx.x * blockDim.x + threadIdx.x;
    if (edge >= E) return;
    int d = dst[edge];
    float ex = __expf(alpha[edge] - funord(mEnc[d]));
    alpha[edge] = ex;                    // reuse buffer: now holds exp()
    atomicAdd(&den[d], ex);
}

// ---------------- attention phase 3: weighted message scatter ---------------
// one wave32 per edge: out[dst] += (ex/denom) * (v[src] + e)
__global__ __launch_bounds__(256)
void attn_scatter(const float* __restrict__ v, const _Float16* __restrict__ e,
                  const int* __restrict__ src, const int* __restrict__ dst,
                  const float* __restrict__ exb, const float* __restrict__ den,
                  float* __restrict__ out, int E, int C)
{
    int edge = blockIdx.x * 8 + (threadIdx.x >> 5);
    int lane = threadIdx.x & 31;
    if (edge >= E) return;
    int s = src[edge], d = dst[edge];
    float a = exb[edge] / (den[d] + 1e-16f);
    for (int c = lane; c < C; c += 32)
        atomicAdd(&out[(size_t)d * C + c],
                  a * (v[(size_t)s * C + c] + (float)e[(size_t)edge * C + c]));
}

// ---------------- classifier: out[n] = relu(h[n,:]) . Wc + bc ---------------
__global__ void classify(const float* __restrict__ h, const float* __restrict__ Wc,
                         const float* __restrict__ bc, float* __restrict__ out,
                         int n, int C)
{
    int i = blockIdx.x * blockDim.x + threadIdx.x;
    if (i >= n) return;
    float acc = bc[0];
    for (int c = 0; c < C; ++c) {
        float v = h[(size_t)i * C + c];
        v = v > 0.0f ? v : 0.0f;
        acc += v * Wc[c];
    }
    out[i] = acc;
}

// ---------------------------------------------------------------------------
extern "C" void kernel_launch(void* const* d_in, const int* in_sizes, int n_in,
                              void* d_out, int out_size, void* d_ws, size_t ws_size,
                              hipStream_t stream)
{
    (void)in_sizes; (void)n_in; (void)out_size; (void)ws_size;

    const float* x   = (const float*)d_in[0];
    const int*   ei  = (const int*)d_in[1];
    const float* ea  = (const float*)d_in[2];
    const float* Wq1 = (const float*)d_in[3];  const float* bq1 = (const float*)d_in[4];
    const float* Wk1 = (const float*)d_in[5];  const float* bk1 = (const float*)d_in[6];
    const float* Wv1 = (const float*)d_in[7];  const float* bv1 = (const float*)d_in[8];
    const float* We1 = (const float*)d_in[9];
    const float* Ws1 = (const float*)d_in[10]; const float* bs1 = (const float*)d_in[11];
    const float* Wq2 = (const float*)d_in[12]; const float* bq2 = (const float*)d_in[13];
    const float* Wk2 = (const float*)d_in[14]; const float* bk2 = (const float*)d_in[15];
    const float* Wv2 = (const float*)d_in[16]; const float* bv2 = (const float*)d_in[17];
    const float* We2 = (const float*)d_in[18];
    const float* Ws2 = (const float*)d_in[19]; const float* bs2 = (const float*)d_in[20];
    const float* Wc  = (const float*)d_in[21]; const float* bc  = (const float*)d_in[22];

    const int* srcIdx = ei;
    const int* dstIdx = ei + N_EDGES;

    // -------- workspace carve-out --------
    char* base = (char*)d_ws;
    size_t off = 0;
    auto alloc = [&](size_t bytes) -> void* {
        void* p = base + off;
        off = (off + bytes + 255) & ~(size_t)255;
        return p;
    };
    _Float16* xh   = (_Float16*)alloc((size_t)N_NODES * DIN * 2);
    _Float16* eah  = (_Float16*)alloc((size_t)N_EDGES * DE  * 2);
    // transposed f16 weights: [N][K]
    _Float16* wq1t = (_Float16*)alloc((size_t)DIN * H1 * 2);
    _Float16* wk1t = (_Float16*)alloc((size_t)DIN * H1 * 2);
    _Float16* wv1t = (_Float16*)alloc((size_t)DIN * H1 * 2);
    _Float16* we1t = (_Float16*)alloc((size_t)DE  * H1 * 2);
    _Float16* ws1t = (_Float16*)alloc((size_t)DIN * H1 * 2);
    _Float16* wq2t = (_Float16*)alloc((size_t)H1 * H2 * 2);
    _Float16* wk2t = (_Float16*)alloc((size_t)H1 * H2 * 2);
    _Float16* wv2t = (_Float16*)alloc((size_t)H1 * H2 * 2);
    _Float16* we2t = (_Float16*)alloc((size_t)DE * H2 * 2);
    _Float16* ws2t = (_Float16*)alloc((size_t)H1 * H2 * 2);
    float* q1   = (float*)alloc((size_t)N_NODES * H1 * 4);
    float* k1   = (float*)alloc((size_t)N_NODES * H1 * 4);
    float* v1   = (float*)alloc((size_t)N_NODES * H1 * 4);
    float* agg1 = (float*)alloc((size_t)N_NODES * H1 * 4);
    _Float16* eh = (_Float16*)alloc((size_t)N_EDGES * H1 * 2);  // reused for layer 2
    unsigned* mEnc = (unsigned*)alloc((size_t)N_NODES * 4);
    float*    den  = (float*)   alloc((size_t)N_NODES * 4);
    float*    exb  = (float*)   alloc((size_t)N_EDGES * 4);
    _Float16* h1h  = (_Float16*)alloc((size_t)N_NODES * H1 * 2);
    float* q2   = (float*)alloc((size_t)N_NODES * H2 * 4);
    float* k2   = (float*)alloc((size_t)N_NODES * H2 * 4);
    float* v2   = (float*)alloc((size_t)N_NODES * H2 * 4);
    float* agg2 = (float*)alloc((size_t)N_NODES * H2 * 4);

    auto cvt = [&](const float* s, _Float16* d, int n) {
        cvt_f32_f16<<<(n + 255) / 256, 256, 0, stream>>>(s, d, n);
    };
    auto cvtT = [&](const float* s, _Float16* d, int K, int N) {
        cvt_f32_f16_T<<<(K * N + 255) / 256, 256, 0, stream>>>(s, d, K, N);
    };
    auto gemm = [&](const _Float16* A, const _Float16* BT, const float* bias,
                    float* C, _Float16* Ch, int M, int N, int K) {
        dim3 g((unsigned)((M + 63) / 64), (unsigned)(N / 64));
        gemm_f16_wmma<<<g, 128, 0, stream>>>(A, BT, bias, C, Ch, M, N, K);
    };

    // -------- f16 conversions (weights transposed for async B staging) -----
    cvt(x,  xh,  N_NODES * DIN);
    cvt(ea, eah, N_EDGES * DE);
    cvtT(Wq1, wq1t, DIN, H1);  cvtT(Wk1, wk1t, DIN, H1);
    cvtT(Wv1, wv1t, DIN, H1);  cvtT(We1, we1t, DE, H1);
    cvtT(Ws1, ws1t, DIN, H1);
    cvtT(Wq2, wq2t, H1, H2);   cvtT(Wk2, wk2t, H1, H2);
    cvtT(Wv2, wv2t, H1, H2);   cvtT(We2, we2t, DE, H2);
    cvtT(Ws2, ws2t, H1, H2);

    const int ewaves = (N_EDGES + 7) / 8;   // one wave32 per edge, 8 waves/block

    // -------- layer 1: TransformerConv(DIN -> H1) --------
    gemm(xh, wq1t, bq1, q1,   nullptr, N_NODES, H1, DIN);
    gemm(xh, wk1t, bk1, k1,   nullptr, N_NODES, H1, DIN);
    gemm(xh, wv1t, bv1, v1,   nullptr, N_NODES, H1, DIN);
    gemm(xh, ws1t, bs1, agg1, nullptr, N_NODES, H1, DIN);   // skip term seeds output
    gemm(eah, we1t, nullptr, nullptr, eh, N_EDGES, H1, DE); // edge embedding (f16)
    attn_init<<<(N_NODES + 255) / 256, 256, 0, stream>>>(mEnc, den, N_NODES);
    attn_alpha<<<ewaves, 256, 0, stream>>>(q1, k1, eh, srcIdx, dstIdx, exb, mEnc,
                                           N_EDGES, H1, 1.0f / sqrtf((float)H1));
    attn_exp<<<(N_EDGES + 255) / 256, 256, 0, stream>>>(exb, mEnc, den, dstIdx, N_EDGES);
    attn_scatter<<<ewaves, 256, 0, stream>>>(v1, eh, srcIdx, dstIdx, exb, den, agg1,
                                             N_EDGES, H1);
    relu_cvt_f16<<<(N_NODES * H1 + 255) / 256, 256, 0, stream>>>(agg1, h1h, N_NODES * H1);

    // -------- layer 2: TransformerConv(H1 -> H2) --------
    gemm(h1h, wq2t, bq2, q2,   nullptr, N_NODES, H2, H1);
    gemm(h1h, wk2t, bk2, k2,   nullptr, N_NODES, H2, H1);
    gemm(h1h, wv2t, bv2, v2,   nullptr, N_NODES, H2, H1);
    gemm(h1h, ws2t, bs2, agg2, nullptr, N_NODES, H2, H1);
    gemm(eah, we2t, nullptr, nullptr, eh, N_EDGES, H2, DE); // reuse edge-emb buffer
    attn_init<<<(N_NODES + 255) / 256, 256, 0, stream>>>(mEnc, den, N_NODES);
    attn_alpha<<<ewaves, 256, 0, stream>>>(q2, k2, eh, srcIdx, dstIdx, exb, mEnc,
                                           N_EDGES, H2, 1.0f / sqrtf((float)H2));
    attn_exp<<<(N_EDGES + 255) / 256, 256, 0, stream>>>(exb, mEnc, den, dstIdx, N_EDGES);
    attn_scatter<<<ewaves, 256, 0, stream>>>(v2, eh, srcIdx, dstIdx, exb, den, agg2,
                                             N_EDGES, H2);

    // -------- classifier: relu(h2) @ Wc + bc --------
    classify<<<(N_NODES + 255) / 256, 256, 0, stream>>>(agg2, Wc, bc, (float*)d_out,
                                                        N_NODES, H2);
}
